// My_self_attention_63677185131093
// MI455X (gfx1250) — matrix operand order repaired and checked by reading, not compiled, not run on hardware
//
#include <hip/hip_runtime.h>
#include <hip/hip_bf16.h>

#define BB 32
#define LL 2048
#define DD 1024
#define NK (DD / 32)

typedef __bf16 bf16;
typedef __attribute__((ext_vector_type(4)))  __bf16 bf16x4;
typedef __attribute__((ext_vector_type(8)))  __bf16 bf16x8;
typedef __attribute__((ext_vector_type(16))) __bf16 bf16x16;
typedef __attribute__((ext_vector_type(8)))  float  floatx8;
typedef __attribute__((ext_vector_type(4)))  int    i32x4;

#if defined(__HIP_DEVICE_COMPILE__) && __has_builtin(__builtin_amdgcn_global_load_async_to_lds_b128)
#define HAVE_ASYNC 1
#else
#define HAVE_ASYNC 0
#endif

__device__ __forceinline__ bf16 tobf(float f) { return (bf16)f; }

// Async 16-byte global -> LDS copy (ASYNCcnt-tracked); sync fallback otherwise.
__device__ __forceinline__ void async_b128(const bf16* g, bf16* l) {
#if HAVE_ASYNC
  __builtin_amdgcn_global_load_async_to_lds_b128(
      (__attribute__((address_space(1))) i32x4*)(uintptr_t)g,
      (__attribute__((address_space(3))) i32x4*)(uint32_t)(uintptr_t)l,
      0, 0);
#else
  *reinterpret_cast<bf16x8*>(l) = *reinterpret_cast<const bf16x8*>(g);
#endif
}

template <int N>
__device__ __forceinline__ void wait_async() {
#if __has_builtin(__builtin_amdgcn_s_wait_asynccnt)
  __builtin_amdgcn_s_wait_asynccnt(N);
#else
  asm volatile("s_wait_asynccnt %0" ::"n"(N));
#endif
}

__device__ __forceinline__ bf16x16 cat8(bf16x8 lo, bf16x8 hi) {
  return __builtin_shufflevector(lo, hi, 0,1,2,3,4,5,6,7,8,9,10,11,12,13,14,15);
}

// A-fragment (16x32 bf16, M x K). Row-major LDS tile, ldk elements per row.
// Lane l (0-15): row M=l, K = [0..7] then [16..23]; lane l+16: K = [8..15],[24..31].
__device__ __forceinline__ bf16x16 fragA(const bf16* lds, int row0, int ldk, int kbase, int lane) {
  const int half = (lane >> 4) & 1;
  const int m    = lane & 15;
  const bf16* p  = lds + (row0 + m) * ldk + kbase + half * 8;
  bf16x8 lo = *reinterpret_cast<const bf16x8*>(p);
  bf16x8 hi = *reinterpret_cast<const bf16x8*>(p + 16);
  return cat8(lo, hi);
}

// B-fragment (32x16 bf16, K x N) from K-contiguous (transposed) LDS tile T[n][k].
// Lane n (0-15): col N=n, K = [0..7] then [16..23]; lane n+16: K = [8..15],[24..31].
__device__ __forceinline__ bf16x16 fragB(const bf16* lds, int col0, int ldk, int kbase, int lane) {
  const int half = (lane >> 4) & 1;
  const int n    = lane & 15;
  const bf16* p  = lds + (col0 + n) * ldk + kbase + half * 8;
  bf16x8 lo = *reinterpret_cast<const bf16x8*>(p);
  bf16x8 hi = *reinterpret_cast<const bf16x8*>(p + 16);
  return cat8(lo, hi);
}

__device__ __forceinline__ floatx8 wmma_bf16(bf16x16 a, bf16x16 b, floatx8 c) {
  return __builtin_amdgcn_wmma_f32_16x16x32_bf16(false, a, false, b, (short)0, c, false, false);
}

// 128x32 bf16 tile copy, 2 async b128 per thread (256 threads).
__device__ __forceinline__ void copy128x32(const bf16* __restrict__ G, size_t grow0, int k0,
                                           bf16* T, int tid) {
#pragma unroll
  for (int i = 0; i < 2; ++i) {
    const int idx = tid + i * 256;
    const int r = idx >> 2, c = (idx & 3) << 3;
    async_b128(G + (grow0 + r) * DD + k0 + c, &T[r * 32 + c]);
  }
}

// 64x32 bf16 tile copy, 1 async b128 per thread.
__device__ __forceinline__ void copy64x32(const bf16* __restrict__ G, size_t grow0, int k0,
                                          bf16* T, int tid) {
  const int r = tid >> 2, c = (tid & 3) << 3;
  async_b128(G + (grow0 + r) * DD + k0 + c, &T[r * 32 + c]);
}

// ---------------------------------------------------------------------------
// Pass 1: P = x @ W  (fp32 in, bf16 out), for W in {q,k,v} selected by blockIdx.z
// Block tile 128(M) x 128(N); K-step 32. 8 waves: 4 row-groups x 2 col-groups.
// ---------------------------------------------------------------------------
__global__ __launch_bounds__(256)
void proj_kernel(const float* __restrict__ x,
                 const float* __restrict__ wq,
                 const float* __restrict__ wk,
                 const float* __restrict__ wv,
                 bf16* __restrict__ Qb, bf16* __restrict__ Kb, bf16* __restrict__ Vb) {
  const float* W; bf16* O;
  if      (blockIdx.z == 0) { W = wq; O = Qb; }
  else if (blockIdx.z == 1) { W = wk; O = Kb; }
  else                      { W = wv; O = Vb; }

  const int n0   = blockIdx.x * 128;
  const int m0   = blockIdx.y * 128;
  const int tid  = threadIdx.x;
  const int lane = tid & 31;
  const int wave = tid >> 5;
  const int wr = wave >> 1, wc = wave & 1;
  const int half = (lane >> 4) & 1, n = lane & 15;

  __shared__ __align__(16) bf16 Xt[128 * 32];   // row-major [m][k]
  __shared__ __align__(16) bf16 Wt[128 * 32];   // transposed [n][k]

  floatx8 acc[2][4];
#pragma unroll
  for (int i = 0; i < 2; ++i)
#pragma unroll
    for (int j = 0; j < 4; ++j) acc[i][j] = (floatx8)0.0f;

  for (int k0 = 0; k0 < DD; k0 += 32) {
    // x tile 128x32 fp32 -> bf16, coalesced float4 loads
#pragma unroll
    for (int i = 0; i < 4; ++i) {
      const int idx = tid + i * 256;
      const int r = idx >> 3, c = (idx & 7) << 2;
      const float4 v = *reinterpret_cast<const float4*>(x + (size_t)(m0 + r) * DD + k0 + c);
      bf16x4 bv = { tobf(v.x), tobf(v.y), tobf(v.z), tobf(v.w) };
      *reinterpret_cast<bf16x4*>(&Xt[r * 32 + c]) = bv;
    }
    // W tile 32x128 fp32 -> transposed bf16 [n][k]
#pragma unroll
    for (int i = 0; i < 4; ++i) {
      const int idx = tid + i * 256;
      const int r = idx >> 5, c = (idx & 31) << 2;   // r = k row, c = n col
      const float4 v = *reinterpret_cast<const float4*>(W + (size_t)(k0 + r) * DD + n0 + c);
      Wt[(c + 0) * 32 + r] = tobf(v.x);
      Wt[(c + 1) * 32 + r] = tobf(v.y);
      Wt[(c + 2) * 32 + r] = tobf(v.z);
      Wt[(c + 3) * 32 + r] = tobf(v.w);
    }
    __syncthreads();
    bf16x16 af[2], bfr[4];
#pragma unroll
    for (int rt = 0; rt < 2; ++rt) af[rt] = fragA(Xt, wr * 32 + rt * 16, 32, 0, lane);
#pragma unroll
    for (int ct = 0; ct < 4; ++ct) bfr[ct] = fragB(Wt, wc * 64 + ct * 16, 32, 0, lane);
#pragma unroll
    for (int rt = 0; rt < 2; ++rt)
#pragma unroll
      for (int ct = 0; ct < 4; ++ct) acc[rt][ct] = wmma_bf16(af[rt], bfr[ct], acc[rt][ct]);
    __syncthreads();
  }

#pragma unroll
  for (int rt = 0; rt < 2; ++rt)
#pragma unroll
    for (int ct = 0; ct < 4; ++ct) {
      const int col  = n0 + wc * 64 + ct * 16 + n;
      const int rowb = m0 + wr * 32 + rt * 16 + half * 8;
#pragma unroll
      for (int r = 0; r < 8; ++r)
        O[(size_t)(rowb + r) * DD + col] = tobf(acc[rt][ct][r]);
    }
}

// ---------------------------------------------------------------------------
// Pass 2: online column-softmax stats. Block owns (b, 128 key columns m),
// streams over all 16 l-tiles; S tiles via WMMA with double-buffered async
// global->LDS tile loads; per-column running max/sum.
// ---------------------------------------------------------------------------
__global__ __launch_bounds__(256)
void stats_kernel(const bf16* __restrict__ Qb, const bf16* __restrict__ Kb,
                  float* __restrict__ gmax, float* __restrict__ gsum) {
  const int b     = blockIdx.y;
  const int mbase = blockIdx.x * 128;
  const int tid   = threadIdx.x, lane = tid & 31, wave = tid >> 5;
  const int wr = wave >> 1, wc = wave & 1;
  const int n = lane & 15;

  __shared__ __align__(16) bf16 Qt[2][128 * 32];
  __shared__ __align__(16) bf16 Kt[2][128 * 32];   // rows of K : [m][d]
  __shared__ float redA[4][128];
  __shared__ float colmax_s[128];
  __shared__ float rmax_s[128], rsum_s[128];

  if (tid < 128) { rmax_s[tid] = -3.0e38f; rsum_s[tid] = 0.0f; }
  __syncthreads();

  const size_t row0 = (size_t)b * LL;
  const size_t krow = row0 + mbase;

  // prologue prefetch for lt = 0
  copy128x32(Qb, row0 + 0, 0, Qt[0], tid);
  copy128x32(Kb, krow, 0, Kt[0], tid);

  for (int lt = 0; lt < LL / 128; ++lt) {
    const int lbase = lt * 128;
    floatx8 acc[2][4];
#pragma unroll
    for (int i = 0; i < 2; ++i)
#pragma unroll
      for (int j = 0; j < 4; ++j) acc[i][j] = (floatx8)0.0f;

    for (int ks = 0; ks < NK; ++ks) {
      const int cur = ks & 1;
      if (ks + 1 < NK) {
        copy128x32(Qb, row0 + lbase, (ks + 1) * 32, Qt[cur ^ 1], tid);
        copy128x32(Kb, krow, (ks + 1) * 32, Kt[cur ^ 1], tid);
        wait_async<4>();
      } else {
        wait_async<0>();
      }
      __syncthreads();
      bf16x16 af[2], bfr[4];
#pragma unroll
      for (int rt = 0; rt < 2; ++rt) af[rt] = fragA(Qt[cur], wr * 32 + rt * 16, 32, 0, lane);
#pragma unroll
      for (int ct = 0; ct < 4; ++ct) bfr[ct] = fragB(Kt[cur], wc * 64 + ct * 16, 32, 0, lane);
#pragma unroll
      for (int rt = 0; rt < 2; ++rt)
#pragma unroll
        for (int ct = 0; ct < 4; ++ct) acc[rt][ct] = wmma_bf16(af[rt], bfr[ct], acc[rt][ct]);
      __syncthreads();
    }

    // prefetch first tiles of next l-tile; overlaps the reductions below
    if (lt + 1 < LL / 128) {
      copy128x32(Qb, row0 + (lt + 1) * 128, 0, Qt[0], tid);
      copy128x32(Kb, krow, 0, Kt[0], tid);
    }

    // ---- tile column max (reduce over l rows) ----
    float lm[4];
#pragma unroll
    for (int ct = 0; ct < 4; ++ct) {
      float m = -3.0e38f;
#pragma unroll
      for (int rt = 0; rt < 2; ++rt)
#pragma unroll
        for (int r = 0; r < 8; ++r) m = fmaxf(m, acc[rt][ct][r]);
      m = fmaxf(m, __shfl_xor(m, 16, 32));
      lm[ct] = m;
    }
    if (lane < 16)
#pragma unroll
      for (int ct = 0; ct < 4; ++ct) redA[wr][wc * 64 + ct * 16 + n] = lm[ct];
    __syncthreads();
    if (tid < 128)
      colmax_s[tid] = fmaxf(fmaxf(redA[0][tid], redA[1][tid]),
                            fmaxf(redA[2][tid], redA[3][tid]));
    __syncthreads();

    // ---- tile column sum of exp ----
    float ls[4];
#pragma unroll
    for (int ct = 0; ct < 4; ++ct) {
      const float cm = colmax_s[wc * 64 + ct * 16 + n];
      float s = 0.0f;
#pragma unroll
      for (int rt = 0; rt < 2; ++rt)
#pragma unroll
        for (int r = 0; r < 8; ++r) s += __expf(acc[rt][ct][r] - cm);
      s += __shfl_xor(s, 16, 32);
      ls[ct] = s;
    }
    if (lane < 16)
#pragma unroll
      for (int ct = 0; ct < 4; ++ct) redA[wr][wc * 64 + ct * 16 + n] = ls[ct];
    __syncthreads();
    if (tid < 128) {
      const float tsum = redA[0][tid] + redA[1][tid] + redA[2][tid] + redA[3][tid];
      const float tmax = colmax_s[tid];
      const float om = rmax_s[tid];
      const float nm = fmaxf(om, tmax);
      rsum_s[tid] = rsum_s[tid] * __expf(om - nm) + tsum * __expf(tmax - nm);
      rmax_s[tid] = nm;
    }
    __syncthreads();
  }

  if (tid < 128) {
    gmax[(size_t)b * LL + mbase + tid] = rmax_s[tid];
    gsum[(size_t)b * LL + mbase + tid] = rsum_s[tid];
  }
}

// ---------------------------------------------------------------------------
// Pass 3: out[l, d] = sum_m exp(S[l,m]-gmax[m])/gsum[m] * V[m,d].
// Block owns (b, 128 l rows, 128 d cols); loops m in chunks of 64:
// recompute S tile (WMMA, double-buffered async loads), normalize -> bf16 P
// tile in LDS, WMMA P@V.
// ---------------------------------------------------------------------------
__global__ __launch_bounds__(256)
void out_kernel(const bf16* __restrict__ Qb, const bf16* __restrict__ Kb,
                const bf16* __restrict__ Vb,
                const float* __restrict__ gmax, const float* __restrict__ gsum,
                float* __restrict__ out) {
  const int b     = blockIdx.z;
  const int lbase = blockIdx.y * 128;
  const int dbase = blockIdx.x * 128;
  const int tid   = threadIdx.x, lane = tid & 31, wave = tid >> 5;
  const int wr = wave >> 1, wc = wave & 1;
  const int half = (lane >> 4) & 1, n = lane & 15;

  __shared__ __align__(16) bf16 Qt[2][128 * 32];
  __shared__ __align__(16) bf16 Kt2[2][64 * 32];
  __shared__ __align__(16) bf16 Pt[128 * 64];   // [l][m]
  __shared__ __align__(16) bf16 Vt[128 * 64];   // transposed [d][m]

  floatx8 oacc[2][4];
#pragma unroll
  for (int i = 0; i < 2; ++i)
#pragma unroll
    for (int j = 0; j < 4; ++j) oacc[i][j] = (floatx8)0.0f;

  const size_t row0 = (size_t)b * LL;
  const size_t qrow = row0 + lbase;

  // prologue prefetch for mc = 0
  copy128x32(Qb, qrow, 0, Qt[0], tid);
  copy64x32(Kb, row0 + 0, 0, Kt2[0], tid);

  for (int mc = 0; mc < LL / 64; ++mc) {
    const int mbase = mc * 64;

    // ---- S tile: 128(l) x 64(m); wave covers 32l x 32m ----
    floatx8 sacc[2][2];
#pragma unroll
    for (int i = 0; i < 2; ++i)
#pragma unroll
      for (int j = 0; j < 2; ++j) sacc[i][j] = (floatx8)0.0f;

    for (int ks = 0; ks < NK; ++ks) {
      const int cur = ks & 1;
      if (ks + 1 < NK) {
        copy128x32(Qb, qrow, (ks + 1) * 32, Qt[cur ^ 1], tid);
        copy64x32(Kb, row0 + mbase, (ks + 1) * 32, Kt2[cur ^ 1], tid);
        wait_async<3>();
      } else {
        wait_async<0>();
      }
      __syncthreads();
      bf16x16 af[2], bfr[2];
#pragma unroll
      for (int rt = 0; rt < 2; ++rt) af[rt] = fragA(Qt[cur], wr * 32 + rt * 16, 32, 0, lane);
#pragma unroll
      for (int ct = 0; ct < 2; ++ct) bfr[ct] = fragB(Kt2[cur], wc * 32 + ct * 16, 32, 0, lane);
#pragma unroll
      for (int rt = 0; rt < 2; ++rt)
#pragma unroll
        for (int ct = 0; ct < 2; ++ct) sacc[rt][ct] = wmma_bf16(af[rt], bfr[ct], sacc[rt][ct]);
      __syncthreads();
    }

    // prefetch first S tiles of next m-chunk; overlaps normalize + P@V below
    if (mc + 1 < LL / 64) {
      copy128x32(Qb, qrow, 0, Qt[0], tid);
      copy64x32(Kb, row0 + (mc + 1) * 64, 0, Kt2[0], tid);
    }

    // ---- normalize to P (bf16, LDS) ----
#pragma unroll
    for (int rt = 0; rt < 2; ++rt)
#pragma unroll
      for (int ct = 0; ct < 2; ++ct) {
        const int colm = wc * 32 + ct * 16 + n;
        const float cm  = gmax[row0 + mbase + colm];
        const float rcp = 1.0f / gsum[row0 + mbase + colm];
        const int row = wr * 32 + rt * 16 + half * 8;
#pragma unroll
        for (int r = 0; r < 8; ++r)
          Pt[(row + r) * 64 + colm] = tobf(__expf(sacc[rt][ct][r] - cm) * rcp);
      }

    // ---- V chunk 64(m) x 128(d) -> transposed LDS [d][m] ----
#pragma unroll
    for (int i = 0; i < 4; ++i) {
      const int idx = tid + i * 256;
      const int m = idx >> 4, dd = (idx & 15) << 3;
      bf16x8 v = *reinterpret_cast<const bf16x8*>(Vb + (row0 + mbase + m) * DD + dbase + dd);
#pragma unroll
      for (int j = 0; j < 8; ++j) Vt[(dd + j) * 64 + m] = v[j];
    }
    __syncthreads();

    // ---- out += P @ V ----
#pragma unroll
    for (int mk = 0; mk < 64; mk += 32) {
      bf16x16 af[2], bfr[4];
#pragma unroll
      for (int rt = 0; rt < 2; ++rt) af[rt] = fragA(Pt, wr * 32 + rt * 16, 64, mk, lane);
#pragma unroll
      for (int ct = 0; ct < 4; ++ct) bfr[ct] = fragB(Vt, wc * 64 + ct * 16, 64, mk, lane);
#pragma unroll
      for (int rt = 0; rt < 2; ++rt)
#pragma unroll
        for (int ct = 0; ct < 4; ++ct) oacc[rt][ct] = wmma_bf16(af[rt], bfr[ct], oacc[rt][ct]);
    }
    __syncthreads();
  }

#pragma unroll
  for (int rt = 0; rt < 2; ++rt)
#pragma unroll
    for (int ct = 0; ct < 4; ++ct) {
      const int col = dbase + wc * 64 + ct * 16 + n;
      const int row = lbase + wr * 32 + rt * 16 + half * 8;
#pragma unroll
      for (int r = 0; r < 8; ++r)
        out[(row0 + row + r) * DD + col] = oacc[rt][ct][r];
    }
}

extern "C" void kernel_launch(void* const* d_in, const int* in_sizes, int n_in,
                              void* d_out, int out_size, void* d_ws, size_t ws_size,
                              hipStream_t stream) {
  const float* x  = (const float*)d_in[0];
  const float* wq = (const float*)d_in[1];
  const float* wk = (const float*)d_in[2];
  const float* wv = (const float*)d_in[3];
  float* out = (float*)d_out;

  char* ws = (char*)d_ws;
  const size_t projBytes = (size_t)BB * LL * DD * sizeof(bf16);   // 128 MB each
  bf16* Qb = (bf16*)(ws);
  bf16* Kb = (bf16*)(ws + projBytes);
  bf16* Vb = (bf16*)(ws + 2 * projBytes);
  float* gmax = (float*)(ws + 3 * projBytes);
  float* gsum = gmax + (size_t)BB * LL;

  dim3 blk(256);
  proj_kernel<<<dim3(DD / 128, (BB * LL) / 128, 3), blk, 0, stream>>>(x, wq, wk, wv, Qb, Kb, Vb);
  stats_kernel<<<dim3(LL / 128, BB), blk, 0, stream>>>(Qb, Kb, gmax, gsum);
  out_kernel<<<dim3(DD / 128, LL / 128, BB), blk, 0, stream>>>(Qb, Kb, Vb, gmax, gsum, out);
}
